// RNAEncoder_64149631533301
// MI455X (gfx1250) — compile-verified, matrix operand order checked by reading, not run on hardware
//
#include <hip/hip_runtime.h>
#include <hip/hip_fp16.h>

// RGCN (4 layers, mean aggregation per (relation,dst), per-relation GEMM,
// root GEMM, bias, eval BatchNorm, final ReLU) for MI455X / gfx1250.
//
// Memory-bound problem: ~2.3 GB of HBM traffic vs ~25 GFLOP of GEMM.
//  1) count + invert -> inv_cnt[R*N]  (graph-only, once per launch)
//  2) per layer: zero msum, fp32-atomic scatter of h[src] rows into
//     msum[(rel*N+dst)*K], then one fused kernel:
//     sum_r (inv_cnt ⊙ msum_r) @ W_r + h @ root + bias, BN, (ReLU)
//     via v_wmma_f32_16x16x32_f16 with f32 accumulation, software-pipelined
//     (double-buffered LDS A-tile, one barrier per relation).

#define N_NODES  50000
#define N_EDGES  1600000
#define N_REL    20
#define IN_DIM   4
#define HID      64
#define N_LAYERS 4
#define BN_EPS   1e-5f

typedef __attribute__((ext_vector_type(16))) _Float16 v16h;
typedef __attribute__((ext_vector_type(8)))  _Float16 v8h;
typedef __attribute__((ext_vector_type(4)))  _Float16 v4h;
typedef __attribute__((ext_vector_type(8)))  float    v8f;

__device__ __forceinline__ void atomAddF(float* p, float v) {
    // native global_atomic_add_f32 on gfx1250
    unsafeAtomicAdd(p, v);
}

// ---------------------------------------------------------------- counts ----
__global__ void rgcn_count_kernel(const int* __restrict__ ei,
                                  const int* __restrict__ et,
                                  float* __restrict__ cnt) {
    int e = blockIdx.x * 256 + threadIdx.x;
    if (e < N_EDGES) {
        int dst = ei[N_EDGES + e];
        int t   = et[e];
        atomAddF(&cnt[(size_t)t * N_NODES + dst], 1.0f);
    }
}

__global__ void rgcn_invert_kernel(float* __restrict__ c) {
    size_t i = (size_t)blockIdx.x * 256 + threadIdx.x;
    if (i < (size_t)N_REL * N_NODES) c[i] = 1.0f / fmaxf(c[i], 1.0f);
}

// --------------------------------------------------------------- scatter ----
// layer 0: feature dim 4; 4 lanes per edge (8 edges / wave, 64 / block)
__global__ void rgcn_scatter4_kernel(const float* __restrict__ x,
                                     const int* __restrict__ ei,
                                     const int* __restrict__ et,
                                     float* __restrict__ msum) {
    int tid = threadIdx.x;
    int e = blockIdx.x * 64 + (tid >> 2);
    int f = tid & 3;
    if (e < N_EDGES) {
        int src = ei[e], dst = ei[N_EDGES + e], t = et[e];
        atomAddF(&msum[((size_t)t * N_NODES + dst) * IN_DIM + f],
                 x[(size_t)src * IN_DIM + f]);
    }
}

// hidden layers: feature dim 64; one wave per edge, one float2 per lane
__global__ void rgcn_scatter64_kernel(const float* __restrict__ h,
                                      const int* __restrict__ ei,
                                      const int* __restrict__ et,
                                      float* __restrict__ msum) {
    int lane = threadIdx.x & 31;
    int wave = threadIdx.x >> 5;
    int e = blockIdx.x * 8 + wave;
    if (e < N_EDGES) {
        int src = ei[e], dst = ei[N_EDGES + e], t = et[e];
        const float2 v = *(const float2*)(h + (size_t)src * HID + 2 * lane);
        float* mb = msum + ((size_t)t * N_NODES + dst) * HID + 2 * lane;
        atomAddF(mb,     v.x);
        atomAddF(mb + 1, v.y);
    }
}

// --------------------------------------------------- weight pre-packing ----
// Pack W[rel][k][col] (rel==N_REL => root matrix) into exact B-operand lane
// order for v_wmma_f32_16x16x32_f16, so each lane loads 16 contiguous halfs:
//   flat = (((rel*4 + colTile)*KSTEPS + ks)*32 + lane)*16 + j
//   k    = 32*ks + (lane>=16 ? 8 : 0) + j + (j>=8 ? 8 : 0)   (B: N=lane&15)
template<int KIN, int KSTEPS>
__global__ void rgcn_pack_w_kernel(const float* __restrict__ W,
                                   const float* __restrict__ root,
                                   _Float16* __restrict__ Wt) {
    const int total = (N_REL + 1) * 4 * KSTEPS * 32 * 16;
    for (int idx = blockIdx.x * 256 + threadIdx.x; idx < total;
         idx += gridDim.x * 256) {
        int j    = idx & 15;
        int lane = (idx >> 4) & 31;
        int q    = idx >> 9;                 // (rel*4+ct)*KSTEPS + ks
        int ks   = q % KSTEPS;
        int ct   = (q / KSTEPS) & 3;
        int rel  = q / (KSTEPS * 4);
        int koff = (lane >= 16) ? 8 : 0;
        int k    = 32 * ks + koff + j + ((j >= 8) ? 8 : 0);
        int col  = 16 * ct + (lane & 15);
        float v = 0.0f;
        if (k < KIN) {
            v = (rel < N_REL) ? W[((size_t)rel * KIN + k) * HID + col]
                              : root[(size_t)k * HID + col];
        }
        Wt[idx] = (_Float16)v;
    }
}

// ------------------------------------------------- fused GEMM + BN (+relu) --
// Block = 128 threads (4 waves). Block computes a 16-node x 64-col output
// tile; wave w owns columns [16w,16w+16). Loop rel = 0..20 (20 relations +
// root term). Double-buffered LDS A-tile: store -> barrier -> prefetch next
// relation's A into regs -> WMMA. One barrier per relation.
template<int KIN, bool RELU>
__global__ __launch_bounds__(128)
void rgcn_gemm_kernel(const float* __restrict__ msum,
                      const float* __restrict__ invc,
                      const float* __restrict__ hroot,
                      const _Float16* __restrict__ Wt,
                      const float* __restrict__ bias,
                      const float* __restrict__ gamma,
                      const float* __restrict__ beta,
                      const float* __restrict__ rmean,
                      const float* __restrict__ rvar,
                      float* __restrict__ out) {
    constexpr int KSTEPS = (KIN + 31) / 32;   // 1 for KIN=4, 2 for KIN=64
    __shared__ _Float16 tile[2][16 * 64];     // double-buffered A-tile (4 KB)

    const int tid  = threadIdx.x;
    const int lane = tid & 31;
    const int wave = tid >> 5;
    const int node_base = blockIdx.x * 16;

    v8f acc = {};

    if constexpr (KIN == 64) {
        // staging: 128 threads x 8 contiguous halfs = full 16x64 tile
        const int srow = (tid * 8) >> 6;
        const int sk0  = (tid * 8) & 63;
        const int node = node_base + srow;

        auto loadA = [&](int rel) -> v8h {
            const float* srcrow;
            float scale;
            if (rel < N_REL) {
                srcrow = msum + ((size_t)rel * N_NODES + node) * 64;
                scale  = invc[(size_t)rel * N_NODES + node];
            } else {
                srcrow = hroot + (size_t)node * 64;   // root term, no scale
                scale  = 1.0f;
            }
            const float4 a = *(const float4*)(srcrow + sk0);
            const float4 b = *(const float4*)(srcrow + sk0 + 4);
            v8h r;
            r[0] = (_Float16)(a.x * scale); r[1] = (_Float16)(a.y * scale);
            r[2] = (_Float16)(a.z * scale); r[3] = (_Float16)(a.w * scale);
            r[4] = (_Float16)(b.x * scale); r[5] = (_Float16)(b.y * scale);
            r[6] = (_Float16)(b.z * scale); r[7] = (_Float16)(b.w * scale);
            return r;
        };

        v8h r = loadA(0);
        int buf = 0;
        for (int rel = 0; rel <= N_REL; ++rel) {
            *(v8h*)&tile[buf][tid * 8] = r;       // one ds_store_b128
            __syncthreads();
            if (rel < N_REL) r = loadA(rel + 1);  // overlap with WMMA below

            const int m    = lane & 15;
            const int koff = (lane >= 16) ? 8 : 0;
            #pragma unroll
            for (int ks = 0; ks < KSTEPS; ++ks) {
                const _Float16* ap = &tile[buf][m * 64 + ks * 32 + koff];
                v8h lo = *(const v8h*)ap;
                v8h hi = *(const v8h*)(ap + 16);
                v16h a16 = __builtin_shufflevector(lo, hi,
                            0,1,2,3,4,5,6,7,8,9,10,11,12,13,14,15);
                v16h b16 = *(const v16h*)(Wt +
                            (((size_t)(rel * 4 + wave) * KSTEPS + ks) * 32 + lane) * 16);
                acc = __builtin_amdgcn_wmma_f32_16x16x32_f16(
                          false, a16, false, b16, (short)0, acc, false, false);
            }
            buf ^= 1;
        }
    } else {
        // KIN == 4: k>=4 region of the padded tile is loop-invariant zero.
        // Zero both buffers once; per relation only threads 0-15 write the
        // 16x4 live region (one ds_store_b64 each). No in-loop predication
        // beyond a single uniform tid<16 guard.
        {
            v8h z = {};
            *(v8h*)&tile[0][tid * 8] = z;
            *(v8h*)&tile[1][tid * 8] = z;
        }
        __syncthreads();

        const int node = node_base + tid;          // valid for tid < 16
        auto loadA4 = [&](int rel) -> v4h {
            const float* srcrow;
            float scale;
            if (rel < N_REL) {
                srcrow = msum + ((size_t)rel * N_NODES + node) * 4;
                scale  = invc[(size_t)rel * N_NODES + node];
            } else {
                srcrow = hroot + (size_t)node * 4;
                scale  = 1.0f;
            }
            const float4 a = *(const float4*)srcrow;
            v4h r;
            r[0] = (_Float16)(a.x * scale); r[1] = (_Float16)(a.y * scale);
            r[2] = (_Float16)(a.z * scale); r[3] = (_Float16)(a.w * scale);
            return r;
        };

        v4h r4 = {};
        if (tid < 16) r4 = loadA4(0);
        int buf = 0;
        for (int rel = 0; rel <= N_REL; ++rel) {
            if (tid < 16) *(v4h*)&tile[buf][tid * 64] = r4;
            __syncthreads();
            if (rel < N_REL && tid < 16) r4 = loadA4(rel + 1);

            const int m    = lane & 15;
            const int koff = (lane >= 16) ? 8 : 0;
            const _Float16* ap = &tile[buf][m * 64 + koff];
            v8h lo = *(const v8h*)ap;
            v8h hi = *(const v8h*)(ap + 16);
            v16h a16 = __builtin_shufflevector(lo, hi,
                        0,1,2,3,4,5,6,7,8,9,10,11,12,13,14,15);
            v16h b16 = *(const v16h*)(Wt +
                        ((size_t)(rel * 4 + wave) * 32 + lane) * 16);
            acc = __builtin_amdgcn_wmma_f32_16x16x32_f16(
                      false, a16, false, b16, (short)0, acc, false, false);
            buf ^= 1;
        }
    }

    // epilogue: (acc + bias - run_mean) * gamma*rsqrt(var+eps) + beta, relu
    const int col = wave * 16 + (lane & 15);
    const float sc  = gamma[col] * rsqrtf(rvar[col] + BN_EPS);
    const float off = (bias[col] - rmean[col]) * sc + beta[col];
    const int rowoff = (lane >= 16) ? 8 : 0;   // C/D layout: VGPR i -> M=i(+8)
    #pragma unroll
    for (int i = 0; i < 8; ++i) {
        float v = acc[i] * sc + off;
        if (RELU) v = fmaxf(v, 0.0f);
        out[(size_t)(node_base + rowoff + i) * HID + col] = v;
    }
}

// ------------------------------------------------------------------ host ----
extern "C" void kernel_launch(void* const* d_in, const int* in_sizes, int n_in,
                              void* d_out, int out_size, void* d_ws, size_t ws_size,
                              hipStream_t stream) {
    const float* x     = (const float*)d_in[0];
    const int*   ei    = (const int*)d_in[1];
    const int*   et    = (const int*)d_in[2];
    const float* W0    = (const float*)d_in[3];
    const float* root0 = (const float*)d_in[4];
    const float* b0    = (const float*)d_in[5];
    const float* Wh    = (const float*)d_in[6];
    const float* rooth = (const float*)d_in[7];
    const float* bh    = (const float*)d_in[8];
    const float* gamma = (const float*)d_in[9];
    const float* beta  = (const float*)d_in[10];
    const float* rmean = (const float*)d_in[11];
    const float* rvar  = (const float*)d_in[12];

    // workspace layout (all 256B-aligned): ~286 MB total
    char* ws = (char*)d_ws;
    size_t off = 0;
    float* invc = (float*)(ws + off); off += (size_t)N_REL * N_NODES * 4;          // 4 MB
    float* msum = (float*)(ws + off); off += (size_t)N_REL * N_NODES * HID * 4;    // 256 MB
    float* hA   = (float*)(ws + off); off += (size_t)N_NODES * HID * 4;            // 12.8 MB
    float* hB   = (float*)(ws + off); off += (size_t)N_NODES * HID * 4;            // 12.8 MB
    _Float16* Wt = (_Float16*)(ws + off);                                          // 168 KB

    // 1) segment counts -> inverse counts (graph-only, reused by all layers)
    hipMemsetAsync(invc, 0, (size_t)N_REL * N_NODES * 4, stream);
    rgcn_count_kernel<<<(N_EDGES + 255) / 256, 256, 0, stream>>>(ei, et, invc);
    rgcn_invert_kernel<<<((size_t)N_REL * N_NODES + 255) / 256, 256, 0, stream>>>(invc);

    // 2) layer 0: IN=4 -> 64
    hipMemsetAsync(msum, 0, (size_t)N_REL * N_NODES * IN_DIM * 4, stream);
    rgcn_scatter4_kernel<<<(N_EDGES + 63) / 64, 256, 0, stream>>>(x, ei, et, msum);
    rgcn_pack_w_kernel<IN_DIM, 1><<<168, 256, 0, stream>>>(W0, root0, Wt);
    rgcn_gemm_kernel<IN_DIM, false><<<N_NODES / 16, 128, 0, stream>>>(
        msum, invc, x, Wt, b0, gamma, beta, rmean, rvar, hA);

    // 3) layers 1..3: 64 -> 64 (last one writes d_out with ReLU)
    const float* hin = hA;
    for (int L = 1; L < N_LAYERS; ++L) {
        hipMemsetAsync(msum, 0, (size_t)N_REL * N_NODES * HID * 4, stream);
        rgcn_scatter64_kernel<<<(N_EDGES + 7) / 8, 256, 0, stream>>>(hin, ei, et, msum);
        rgcn_pack_w_kernel<HID, 2><<<336, 256, 0, stream>>>(
            Wh + (size_t)(L - 1) * N_REL * HID * HID,
            rooth + (size_t)(L - 1) * HID * HID, Wt);
        if (L == N_LAYERS - 1) {
            rgcn_gemm_kernel<HID, true><<<N_NODES / 16, 128, 0, stream>>>(
                msum, invc, hin, Wt, bh + (size_t)(L - 1) * HID,
                gamma + (size_t)L * HID, beta + (size_t)L * HID,
                rmean + (size_t)L * HID, rvar + (size_t)L * HID, (float*)d_out);
        } else {
            float* hout = (hin == hA) ? hB : hA;
            rgcn_gemm_kernel<HID, false><<<N_NODES / 16, 128, 0, stream>>>(
                msum, invc, hin, Wt, bh + (size_t)(L - 1) * HID,
                gamma + (size_t)L * HID, beta + (size_t)L * HID,
                rmean + (size_t)L * HID, rvar + (size_t)L * HID, hout);
            hin = hout;
        }
    }
}
